// Porn_81415400062997
// MI455X (gfx1250) — matrix operand order repaired and checked by reading, not compiled, not run on hardware
//
#include <hip/hip_runtime.h>

// ---------------------------------------------------------------------------
// MI455X (gfx1250, wave32) forward pass for the CNN+BiLSTM+NRI reference.
// All dense math runs through v_wmma_f32_16x16x32_f16 (f16 in, f32 acc).
// B matrices are kept in [N][K] (original torch) layout so that every lane's
// B fragment is 16 contiguous halves -> ds_load_b128. k_gemm64 stages its
// LDS tiles with gfx1250 async-LDS loads (ASYNCcnt) instead of VGPR bounces.
// Workspace requirement: ~330 MB.
// ---------------------------------------------------------------------------

typedef _Float16 h16;
typedef __attribute__((ext_vector_type(16))) _Float16 v16h;
typedef __attribute__((ext_vector_type(8)))  _Float16 v8h;
typedef __attribute__((ext_vector_type(8)))  float    v8f;

#define EPSV 1e-5f
#define DEV static __device__ __forceinline__

DEV float sigm(float x) { return 1.f / (1.f + __expf(-x)); }

// A fragment (16x32 f16): lane m = lane&15, half = lane>>4.
DEV v16h load_a_frag(const h16* A, int lda, int k0, int lane) {
  union { v16h v; h16 e[16]; } u;
  const int half = (lane >> 4) & 1;
  const int m = lane & 15;
  const h16* row = A + m * lda + k0;
#pragma unroll
  for (int r = 0; r < 8; ++r) {
    int kb = 16*half + 2*r;
    u.e[2*r]   = row[kb];
    u.e[2*r+1] = row[kb+1];
  }
  return u.v;
}

// B fragment from an [N][K]-layout LDS tile: 16 contiguous halves per lane.
DEV v16h load_b_fragT(const h16* Bt, int ldb, int k0, int lane) {
  union { v16h v; h16 e[16]; } u;
  const int half = (lane >> 4) & 1;
  const int n = lane & 15;
  const h16* p = Bt + n * ldb + k0 + 16 * half;
#pragma unroll
  for (int j = 0; j < 16; ++j) u.e[j] = p[j];
  return u.v;
}

DEV v8f wmma32(v16h a, v16h b, v8f c) {
  return __builtin_amdgcn_wmma_f32_16x16x32_f16(false, a, false, b, (short)0, c,
                                                false, false);
}

// gfx1250 async global->LDS 16-byte copy (tracked by ASYNCcnt)
DEV void async_copy_b128(const void* gaddr, void* lds) {
  unsigned l = (unsigned)(size_t)lds;   // low 32 bits of generic = LDS offset
  asm volatile("global_load_async_to_lds_b128 %0, %1, off"
               :: "v"(l), "v"(gaddr) : "memory");
}
DEV void async_wait0() {
  asm volatile("s_wait_asynccnt 0x0" ::: "memory");
}

// ---------------------------------------------------------------------------
// Generic 64x64-tile GEMM: C[M,N] = A[M,K] * Bt[N,K]^T (+bias)(+relu)
// M, N, K multiples of 64. 8 waves, 2 tiles/wave sharing one B fragment.
// LDS tiles are filled by async-LDS DMA loads.
// ---------------------------------------------------------------------------
__global__ __launch_bounds__(256) void k_gemm64(
    const h16* __restrict__ A, int lda,
    const h16* __restrict__ B, int ldb, int K,   // B in [N][K] layout
    const float* __restrict__ bias, int relu,
    h16* __restrict__ Ch, float* __restrict__ Cf, int ldc)
{
  __shared__ h16 As[64][72];
  __shared__ h16 Bs[64][72];   // [n][k]
  const int tid = threadIdx.x, lane = tid & 31, wv = tid >> 5;
  const long m0 = (long)blockIdx.x * 64, n0 = (long)blockIdx.y * 64;
  const int mt0 = wv >> 2, nt0 = wv & 3;
  const int mt1 = mt0 + 2;                  // (wv+8)>>2 ; same nt
  v8f acc0 = {}; v8f acc1 = {};
  for (int k0 = 0; k0 < K; k0 += 64) {
#pragma unroll
    for (int i = 0; i < 2; ++i) {           // 16-byte async staging
      int e = tid + i * 256;
      int r = e >> 3, c = (e & 7) << 3;
      async_copy_b128(&A[(m0 + r) * (long)lda + k0 + c], &As[r][c]);
      async_copy_b128(&B[(long)(n0 + r) * ldb + k0 + c], &Bs[r][c]);
    }
    if (k0 + 64 < K) {   // gfx1250 global_prefetch_b8 on the next K-chunk
      __builtin_prefetch(&A[(m0 + (tid >> 6)) * (long)lda + k0 + 64 + (tid & 63)], 0, 1);
      __builtin_prefetch(&B[(long)(n0 + (tid >> 6)) * ldb + k0 + 64 + (tid & 63)], 0, 1);
    }
    async_wait0();
    __syncthreads();
#pragma unroll
    for (int kk = 0; kk < 64; kk += 32) {
      v16h bf = load_b_fragT(&Bs[nt0*16][0], 72, kk, lane);
      acc0 = wmma32(load_a_frag(&As[mt0*16][0], 72, kk, lane), bf, acc0);
      acc1 = wmma32(load_a_frag(&As[mt1*16][0], 72, kk, lane), bf, acc1);
    }
    __syncthreads();
  }
  const int n = lane & 15, mo = (lane >> 4) ? 8 : 0;
  auto emit = [&](v8f acc, int mt, int nt) {
    union { v8f v; float e[8]; } u; u.v = acc;
    long gc = n0 + nt * 16 + n;
    float bv = bias ? bias[gc] : 0.f;
#pragma unroll
    for (int v = 0; v < 8; ++v) {
      long gr = m0 + mt * 16 + v + mo;
      float x = u.e[v] + bv;
      if (relu) x = x > 0.f ? x : 0.f;
      if (Ch) Ch[gr * ldc + gc] = (h16)x;
      if (Cf) Cf[gr * ldc + gc] = x;
    }
  };
  emit(acc0, mt0, nt0);
  emit(acc1, mt1, nt0);
}

// ---------------------------------------------------------------------------
// conv1 (5x9, 1->64) implicit GEMM: 64 positions x K=45(pad64).
// (conv bias omitted: cancels in the following training-mode BN)
// ---------------------------------------------------------------------------
__global__ __launch_bounds__(256) void k_conv1(
    const float* __restrict__ in,  // [5120][5][200]
    const h16* __restrict__ W,     // [64][64]  [c][k] (padded)
    h16* __restrict__ Y)           // [5120][64][192]
{
  __shared__ h16 As[64][72];
  __shared__ h16 Bs[64][72];
  const int tid = threadIdx.x, lane = tid & 31, wv = tid >> 5;
  const int s = blockIdx.x, t0 = blockIdx.y * 64;
  const float* src = in + (long)s * 1000;
#pragma unroll
  for (int i = 0; i < 2; ++i) {
    int e = tid + i * 256;
    int r = e >> 3, c0 = (e & 7) << 3;
    *(v8h*)&Bs[r][c0] = *(const v8h*)&W[r * 64 + c0];
    union { v8h v; h16 e8[8]; } t;
#pragma unroll
    for (int j = 0; j < 8; ++j) {
      int k = c0 + j;
      t.e8[j] = (k < 45) ? (h16)src[(k / 9) * 200 + t0 + r + (k % 9)] : (h16)0.f;
    }
    *(v8h*)&As[r][c0] = t.v;
  }
  __syncthreads();
  const int mt0 = wv >> 2, nt0 = wv & 3, mt1 = mt0 + 2;
  v8f acc0 = {}, acc1 = {};
#pragma unroll
  for (int kk = 0; kk < 64; kk += 32) {
    v16h bf = load_b_fragT(&Bs[nt0*16][0], 72, kk, lane);
    acc0 = wmma32(load_a_frag(&As[mt0*16][0], 72, kk, lane), bf, acc0);
    acc1 = wmma32(load_a_frag(&As[mt1*16][0], 72, kk, lane), bf, acc1);
  }
  const int n = lane & 15, mo = (lane >> 4) ? 8 : 0;
  auto emit = [&](v8f acc, int mt, int nt) {
    union { v8f v; float e[8]; } u; u.v = acc;
    int gc = nt * 16 + n;
#pragma unroll
    for (int v = 0; v < 8; ++v) {
      int gr = t0 + mt * 16 + v + mo;
      Y[((long)s * 64 + gc) * 192 + gr] = (h16)u.e[v];
    }
  };
  emit(acc0, mt0, nt0); emit(acc1, mt1, nt0);
}

// ---------------------------------------------------------------------------
// conv2/conv3 (k=5, 64->64), K = 320 in 5 chunks. W in [co][320] layout.
// ---------------------------------------------------------------------------
__global__ __launch_bounds__(256) void k_conv23(
    const h16* __restrict__ X,   // [5120][64][Lin]
    const h16* __restrict__ W,   // [64][320]
    h16* __restrict__ Y,         // [5120][64][Lout]
    int Lin, int Lout)
{
  __shared__ h16 As[64][72];
  __shared__ h16 Bs[64][72];
  const int tid = threadIdx.x, lane = tid & 31, wv = tid >> 5;
  const int s = blockIdx.x, t0 = blockIdx.y * 64;
  const int mt0 = wv >> 2, nt0 = wv & 3, mt1 = mt0 + 2;
  v8f acc0 = {}, acc1 = {};
  for (int k0 = 0; k0 < 320; k0 += 64) {
#pragma unroll
    for (int i = 0; i < 2; ++i) {
      int e = tid + i * 256;
      int r = e >> 3, c0 = (e & 7) << 3;
      *(v8h*)&Bs[r][c0] = *(const v8h*)&W[r * 320 + k0 + c0];
      union { v8h v; h16 e8[8]; } t;
      int tpos = t0 + r;
#pragma unroll
      for (int j = 0; j < 8; ++j) {
        int k = k0 + c0 + j, ci = k / 5, w = k % 5;
        t.e8[j] = (tpos < Lout) ? X[((long)s * 64 + ci) * Lin + tpos + w] : (h16)0.f;
      }
      *(v8h*)&As[r][c0] = t.v;
    }
    __syncthreads();
#pragma unroll
    for (int kk = 0; kk < 64; kk += 32) {
      v16h bf = load_b_fragT(&Bs[nt0*16][0], 72, kk, lane);
      acc0 = wmma32(load_a_frag(&As[mt0*16][0], 72, kk, lane), bf, acc0);
      acc1 = wmma32(load_a_frag(&As[mt1*16][0], 72, kk, lane), bf, acc1);
    }
    __syncthreads();
  }
  const int n = lane & 15, mo = (lane >> 4) ? 8 : 0;
  auto emit = [&](v8f acc, int mt, int nt) {
    union { v8f v; float e[8]; } u; u.v = acc;
    int gc = nt * 16 + n;
#pragma unroll
    for (int v = 0; v < 8; ++v) {
      int gr = t0 + mt * 16 + v + mo;
      if (gr < Lout) Y[((long)s * 64 + gc) * Lout + gr] = (h16)u.e[v];
    }
  };
  emit(acc0, mt0, nt0); emit(acc1, mt1, nt0);
}

// ---------------------------------------------------------------------------
// per-channel BN stats (sum, sumsq) over [S][64][L]
// ---------------------------------------------------------------------------
template <typename T>
__global__ __launch_bounds__(256) void k_stats_ch(
    const T* __restrict__ x, int S, int L, float* __restrict__ sums)
{
  __shared__ float ls[64], lq[64];
  const int tid = threadIdx.x;
  if (tid < 64) { ls[tid] = 0.f; lq[tid] = 0.f; }
  __syncthreads();
  const int rows = S * 64;
  for (int row = blockIdx.x * 256 + tid; row < rows; row += gridDim.x * 256) {
    const T* p = x + (long)row * L;
    float s0 = 0.f, s1 = 0.f;
    for (int t = 0; t < L; ++t) { float v = (float)p[t]; s0 += v; s1 += v * v; }
    atomicAdd(&ls[row & 63], s0); atomicAdd(&lq[row & 63], s1);
  }
  __syncthreads();
  if (tid < 64) { atomicAdd(&sums[tid*2], ls[tid]); atomicAdd(&sums[tid*2+1], lq[tid]); }
}

// BN (channel) + relu + maxpool2; optionally emit time-major [t][S][64]
__global__ __launch_bounds__(256) void k_bn_pool(
    const h16* __restrict__ Y, int S, int L,
    const float* __restrict__ sums, float cnt,
    const float* __restrict__ g, const float* __restrict__ be,
    h16* __restrict__ X, int timeMajor)
{
  const int Lh = L >> 1;
  const long total = (long)S * 64 * Lh;
  for (long i = (long)blockIdx.x * 256 + threadIdx.x; i < total; i += (long)gridDim.x * 256) {
    int t = (int)(i % Lh); long rc = i / Lh; int c = (int)(rc & 63); int s = (int)(rc >> 6);
    float mu = sums[c*2] / cnt;
    float var = sums[c*2+1] / cnt - mu * mu;
    float sc = g[c] * rsqrtf(var + EPSV);
    float v0 = ((float)Y[rc * L + 2*t]     - mu) * sc + be[c];
    float v1 = ((float)Y[rc * L + 2*t + 1] - mu) * sc + be[c];
    v0 = v0 > 0.f ? v0 : 0.f; v1 = v1 > 0.f ? v1 : 0.f;
    float m = v0 > v1 ? v0 : v1;
    if (timeMajor) X[((long)t * S + s) * 64 + c] = (h16)m;
    else           X[rc * Lh + t] = (h16)m;
  }
}

// ---------------------------------------------------------------------------
// LSTM recurrent scan: one block owns 32 batch rows; h,c resident in LDS.
// gates = h @ Whh^T (WMMA, Whh kept in original [256][64] layout) + xproj[t].
// ---------------------------------------------------------------------------
__global__ __launch_bounds__(256) void k_lstm(
    const h16* __restrict__ xp,   // [21][5120][256] (includes gate bias)
    const h16* __restrict__ whh,  // [256][64]  (plain torch layout)
    h16* __restrict__ hbuf,       // [21][5120][128]
    int reverse, int dirofs)
{
  __shared__ h16  hsm[32][72];
  __shared__ h16  wsm[256][64];   // [n][k]
  __shared__ float csm[32][64];
  __shared__ h16  gsm[32][264];
  const int tid = threadIdx.x, lane = tid & 31, wv = tid >> 5;
  const int row0 = blockIdx.x * 32;
#pragma unroll
  for (int i = 0; i < 8; ++i) {           // 2048 16-byte chunks
    int e = tid + i * 256;
    int nn = e >> 3, c0 = (e & 7) << 3;
    *(v8h*)&wsm[nn][c0] = *(const v8h*)&whh[nn * 64 + c0];
  }
#pragma unroll
  for (int i = 0; i < 8; ++i) { int e = tid + i * 256; hsm[e >> 6][e & 63] = (h16)0.f; csm[e >> 6][e & 63] = 0.f; }
  __syncthreads();
  const int mt = wv >> 2;                 // wave's M tile (constant)
  for (int step = 0; step < 21; ++step) {
    const int t = reverse ? (20 - step) : step;
    const long xbase = ((long)t * 5120 + row0) * 256;
    v16h a0 = load_a_frag(&hsm[mt*16][0], 72, 0, lane);
    v16h a1 = load_a_frag(&hsm[mt*16][0], 72, 32, lane);
#pragma unroll
    for (int i = 0; i < 4; ++i) {         // 32 tiles (2x16) / 8 waves
      int nt = (4 * (wv & 3) + i);
      v8f acc = {};
      acc = wmma32(a0, load_b_fragT(&wsm[nt*16][0], 64, 0, lane), acc);
      acc = wmma32(a1, load_b_fragT(&wsm[nt*16][0], 64, 32, lane), acc);
      union { v8f v; float e[8]; } u; u.v = acc;
      const int n = lane & 15, mo = (lane >> 4) ? 8 : 0;
#pragma unroll
      for (int v = 0; v < 8; ++v) {
        int r = mt * 16 + v + mo, c = nt * 16 + n;
        gsm[r][c] = (h16)(u.e[v] + (float)xp[xbase + (long)r * 256 + c]);
      }
    }
    __syncthreads();
#pragma unroll
    for (int i = 0; i < 8; ++i) {
      int e = tid + i * 256; int r = e >> 6, n = e & 63;
      float ig = (float)gsm[r][n],        fg = (float)gsm[r][64 + n];
      float gg = (float)gsm[r][128 + n],  og = (float)gsm[r][192 + n];
      float c = sigm(fg) * csm[r][n] + sigm(ig) * tanhf(gg);
      float h = sigm(og) * tanhf(c);
      csm[r][n] = c; hsm[r][n] = (h16)h;
      hbuf[((long)t * 5120 + row0 + r) * 128 + dirofs + n] = (h16)h;
    }
    __syncthreads();
  }
}

// --------------------------- small utility kernels -------------------------

__global__ void k_cast(h16* dst, const float* src, int n) {
  for (int i = blockIdx.x * 256 + threadIdx.x; i < n; i += gridDim.x * 256)
    dst[i] = (h16)src[i];
}

__global__ void k_cast_conv1(h16* dst, const float* src) {   // [c][k] pad 45->64
  for (int i = blockIdx.x * 256 + threadIdx.x; i < 4096; i += gridDim.x * 256) {
    int c = i >> 6, k = i & 63;
    dst[i] = (k < 45) ? (h16)src[c * 45 + k] : (h16)0.f;
  }
}

__global__ void k_edge_idx(const float* rr, const float* rs, int* se, int* re) {
  int e = threadIdx.x;
  if (e < 20) {
    int si = 0, ri = 0;
    for (int n = 0; n < 5; ++n) {
      if (rs[e * 5 + n] > 0.5f) si = n;
      if (rr[e * 5 + n] > 0.5f) ri = n;
    }
    se[e] = si; re[e] = ri;
  }
}

__global__ void k_node2edge(const h16* __restrict__ src, int F,
                            const int* __restrict__ se, const int* __restrict__ re,
                            h16* __restrict__ dst, int ldd, int off)
{
  const long total = 1024l * 20 * (2 * F);
  for (long i = (long)blockIdx.x * 256 + threadIdx.x; i < total; i += (long)gridDim.x * 256) {
    int k = (int)(i % (2 * F));
    long be_ = i / (2 * F);
    int e = (int)(be_ % 20);
    long b = be_ / 20;
    int node = (k < F) ? se[e] : re[e];
    int kk = (k < F) ? k : k - F;
    dst[(b * 20 + e) * ldd + off + k] = src[(b * 5 + node) * F + kk];
  }
}

__global__ void k_copy_rows(const h16* __restrict__ src, int F,
                            h16* __restrict__ dst, int ldd, int off, long rows)
{
  const long total = rows * F;
  for (long i = (long)blockIdx.x * 256 + threadIdx.x; i < total; i += (long)gridDim.x * 256) {
    int k = (int)(i % F); long r = i / F;
    dst[r * ldd + off + k] = src[r * F + k];
  }
}

__global__ void k_edge2node(const h16* __restrict__ src, int F,
                            const int* __restrict__ re, h16* __restrict__ dst)
{
  const long total = 5120l * F;
  for (long i = (long)blockIdx.x * 256 + threadIdx.x; i < total; i += (long)gridDim.x * 256) {
    int k = (int)(i % F); long s = i / F; int n = (int)(s % 5); long b = s / 5;
    float acc = 0.f;
    for (int e = 0; e < 20; ++e)
      if (re[e] == n) acc += (float)src[(b * 20 + e) * F + k];
    dst[i] = (h16)(acc * 0.2f);
  }
}

__global__ __launch_bounds__(256) void k_stats_feat(
    const h16* __restrict__ x, long rows, int F, float* __restrict__ sums)
{
  int f = threadIdx.x;
  if (f >= F) return;
  float s0 = 0.f, s1 = 0.f;
  for (long r = blockIdx.x; r < rows; r += gridDim.x) {
    float v = (float)x[r * F + f]; s0 += v; s1 += v * v;
  }
  atomicAdd(&sums[f * 2], s0); atomicAdd(&sums[f * 2 + 1], s1);
}

__global__ void k_bn_feat(const h16* __restrict__ raw, long rows, int F,
                          const float* __restrict__ sums, float cnt,
                          const float* __restrict__ g, const float* __restrict__ be,
                          h16* __restrict__ out)
{
  const long total = rows * F;
  for (long i = (long)blockIdx.x * 256 + threadIdx.x; i < total; i += (long)gridDim.x * 256) {
    int f = (int)(i % F);
    float mu = sums[f * 2] / cnt, var = sums[f * 2 + 1] / cnt - mu * mu;
    float v = ((float)raw[i] - mu) * rsqrtf(var + EPSV) * g[f] + be[f];
    out[i] = (h16)(v > 0.f ? v : 0.f);
  }
}

__global__ void k_logits(const h16* __restrict__ x, const float* __restrict__ w,
                         const float* __restrict__ bo, float* __restrict__ relt)
{
  int row = blockIdx.x * 256 + threadIdx.x;
  if (row >= 20480) return;
  float l0 = bo[0], l1 = bo[1];
  for (int k = 0; k < 64; ++k) {
    float v = (float)x[(long)row * 64 + k];
    l0 += v * w[k]; l1 += v * w[64 + k];
  }
  float mx = l0 > l1 ? l0 : l1;
  float e0 = __expf(l0 - mx), e1 = __expf(l1 - mx), s = e0 + e1;
  relt[row * 2] = e0 / s; relt[row * 2 + 1] = e1 / s;
}

__global__ void k_aggregate(const h16* __restrict__ m0, const h16* __restrict__ m1,
                            const float* __restrict__ relt, const int* __restrict__ re,
                            float* __restrict__ nod)
{
  const long total = 1024l * 5 * 512;
  for (long i = (long)blockIdx.x * 256 + threadIdx.x; i < total; i += (long)gridDim.x * 256) {
    int f = (int)(i % 512); long s = i / 512; int n = (int)(s % 5); long b = s / 5;
    int part = f >> 8, k = f & 255;
    const h16* m = part ? m1 : m0;
    float acc = 0.f;
    for (int e = 0; e < 20; ++e)
      if (re[e] == n) {
        long row = b * 20 + e;
        acc += (float)m[row * 256 + k] * relt[row * 2 + part];
      }
    nod[i] = acc;
  }
}

__global__ __launch_bounds__(256) void k_stats_scalar(
    const float* __restrict__ x, long n, float* __restrict__ sums)
{
  float s0 = 0.f, s1 = 0.f;
  for (long i = (long)blockIdx.x * 256 + threadIdx.x; i < n; i += (long)gridDim.x * 256) {
    float v = x[i]; s0 += v; s1 += v * v;
  }
  __shared__ float r0[256], r1[256];
  r0[threadIdx.x] = s0; r1[threadIdx.x] = s1; __syncthreads();
  for (int s = 128; s > 0; s >>= 1) {
    if (threadIdx.x < s) { r0[threadIdx.x] += r0[threadIdx.x + s]; r1[threadIdx.x] += r1[threadIdx.x + s]; }
    __syncthreads();
  }
  if (threadIdx.x == 0) { atomicAdd(&sums[0], r0[0]); atomicAdd(&sums[1], r1[0]); }
}

// ---------------------------------------------------------------------------
// Decoder: bn6(agg) = per-channel affine of nod's scalar stats; expand the
// relu'd A tile in LDS, conv7 as WMMA GEMM (M=b*f, K=320, N=64); K/V fused.
// ---------------------------------------------------------------------------
__global__ __launch_bounds__(256) void k_conv7(
    const float* __restrict__ nod,  // [1024][5][512]
    const float* __restrict__ st,   // scalar stats of nod
    const float* __restrict__ w6, const float* __restrict__ g6, const float* __restrict__ be6,
    const h16* __restrict__ W7,     // [64][320]  (plain torch layout)
    const float* __restrict__ Kw, const float* __restrict__ Vw,
    float* __restrict__ merge,      // [1024][64][512]
    float* __restrict__ kraw, float* __restrict__ vraw,  // [1024][5][512]
    float cnt)
{
  __shared__ h16 As[64][328];
  __shared__ h16 Bs[64][72];      // [n][k] chunk
  __shared__ float sc[64], sb[64];
  const int tid = threadIdx.x, lane = tid & 31, wv = tid >> 5;
  const int blk = blockIdx.x;
  const int b = blk >> 3, f0 = (blk & 7) * 64;
  const float mu = st[0] / cnt;
  const float var = st[1] / cnt - mu * mu;
  if (tid < 64) {
    float w = w6[tid];
    sc[tid] = g6[tid] * w * rsqrtf(w * w * var + EPSV);
    sb[tid] = be6[tid];
  }
  __syncthreads();
#pragma unroll
  for (int i = 0; i < 10; ++i) {          // 2560 16-byte chunks
    int e = tid + i * 256;
    int r = e / 40, c0 = (e % 40) * 8;
    union { v8h v; h16 e8[8]; } t;
#pragma unroll
    for (int j = 0; j < 8; ++j) {
      int k = c0 + j, c = k / 5, n = k % 5;
      float v = nod[((long)b * 5 + n) * 512 + f0 + r];
      float a = sc[c] * (v - mu) + sb[c];
      t.e8[j] = (h16)(a > 0.f ? a : 0.f);
    }
    *(v8h*)&As[r][c0] = t.v;
  }
  __syncthreads();
  for (int e = tid; e < 320; e += 256) {  // fused K/V 1x1 convs
    int r = e / 5, n = e % 5;
    float sk = 0.f, sv = 0.f;
    for (int c = 0; c < 64; ++c) {
      float a = (float)As[r][c * 5 + n];
      sk += Kw[c] * a; sv += Vw[c] * a;
    }
    kraw[((long)b * 5 + n) * 512 + f0 + r] = sk;
    vraw[((long)b * 5 + n) * 512 + f0 + r] = sv;
  }
  const int mt0 = wv >> 2, nt0 = wv & 3, mt1 = mt0 + 2;
  v8f acc0 = {}, acc1 = {};
  for (int k0 = 0; k0 < 320; k0 += 64) {
    __syncthreads();
#pragma unroll
    for (int i = 0; i < 2; ++i) {
      int e = tid + i * 256;
      int r = e >> 3, c0 = (e & 7) << 3;
      *(v8h*)&Bs[r][c0] = *(const v8h*)&W7[r * 320 + k0 + c0];
    }
    __syncthreads();
#pragma unroll
    for (int kk = 0; kk < 64; kk += 32) {
      v16h bf = load_b_fragT(&Bs[nt0*16][0], 72, kk, lane);
      acc0 = wmma32(load_a_frag(&As[mt0*16][0], 328, k0 + kk, lane), bf, acc0);
      acc1 = wmma32(load_a_frag(&As[mt1*16][0], 328, k0 + kk, lane), bf, acc1);
    }
  }
  const int n = lane & 15, mo = (lane >> 4) ? 8 : 0;
  auto emit = [&](v8f acc, int mt, int nt) {
    union { v8f v; float e[8]; } u; u.v = acc;
    int gc = nt * 16 + n;
#pragma unroll
    for (int v = 0; v < 8; ++v) {
      int gr = mt * 16 + v + mo;
      merge[((long)b * 64 + gc) * 512 + f0 + gr] = u.e[v];
    }
  };
  emit(acc0, mt0, nt0); emit(acc1, mt1, nt0);
}

__global__ void k_q(const float* __restrict__ merge, const float* __restrict__ st7,
                    const float* __restrict__ g7, const float* __restrict__ be7,
                    const float* __restrict__ Qw, float* __restrict__ qraw, float cnt)
{
  const long total = 1024l * 512;
  for (long i = (long)blockIdx.x * 256 + threadIdx.x; i < total; i += (long)gridDim.x * 256) {
    int b = (int)(i >> 9), f = (int)(i & 511);
    float s = 0.f;
    for (int o = 0; o < 64; ++o) {
      float mu = st7[o * 2] / cnt;
      float var = st7[o * 2 + 1] / cnt - mu * mu;
      float v = (merge[((long)b * 64 + o) * 512 + f] - mu) * rsqrtf(var + EPSV) * g7[o] + be7[o];
      v = v > 0.f ? v : 0.f;
      s += Qw[o] * v;
    }
    qraw[i] = s;
  }
}

__global__ __launch_bounds__(256) void k_attn(
    const float* __restrict__ qraw, const float* __restrict__ kraw, const float* __restrict__ vraw,
    const float* __restrict__ stQ, const float* __restrict__ stK, const float* __restrict__ stV,
    const float* __restrict__ gQ, const float* __restrict__ bQ,
    const float* __restrict__ gK, const float* __restrict__ bK,
    const float* __restrict__ gV, const float* __restrict__ bV,
    const float* __restrict__ fcw, const float* __restrict__ fcb,
    float* __restrict__ out, float cntQ, float cntKV)
{
  __shared__ float q[512], wq[512], red[256], sco[5];
  const int b = blockIdx.x, tid = threadIdx.x;
  const float muQ = stQ[0] / cntQ, rsQ = rsqrtf(stQ[1] / cntQ - muQ * muQ + EPSV);
  const float muK = stK[0] / cntKV, rsK = rsqrtf(stK[1] / cntKV - muK * muK + EPSV);
  const float muV = stV[0] / cntKV, rsV = rsqrtf(stV[1] / cntKV - muV * muV + EPSV);
  for (int f = tid; f < 512; f += 256) {
    float v = (qraw[(long)b * 512 + f] - muQ) * rsQ * gQ[0] + bQ[0];
    q[f] = v > 0.f ? v : 0.f;
  }
  __syncthreads();
  for (int n = 0; n < 5; ++n) {
    float p = 0.f;
    for (int f = tid; f < 512; f += 256) {
      float kv = (kraw[((long)b * 5 + n) * 512 + f] - muK) * rsK * gK[0] + bK[0];
      kv = kv > 0.f ? kv : 0.f;
      p += kv * q[f];
    }
    red[tid] = p; __syncthreads();
    for (int s = 128; s > 0; s >>= 1) { if (tid < s) red[tid] += red[tid + s]; __syncthreads(); }
    if (tid == 0) sco[n] = red[0] * 0.044194173824159216f;  // 1/sqrt(512)
    __syncthreads();
  }
  if (tid == 0) {
    float mx = sco[0];
    for (int n = 1; n < 5; ++n) mx = sco[n] > mx ? sco[n] : mx;
    float ss = 0.f;
    for (int n = 0; n < 5; ++n) { sco[n] = __expf(sco[n] - mx); ss += sco[n]; }
    for (int n = 0; n < 5; ++n) sco[n] /= ss;
  }
  __syncthreads();
  for (int f = tid; f < 512; f += 256) {
    float acc = q[f];
    for (int n = 0; n < 5; ++n) {
      float vv = (vraw[((long)b * 5 + n) * 512 + f] - muV) * rsV * gV[0] + bV[0];
      vv = vv > 0.f ? vv : 0.f;
      acc += vv * sco[n];
    }
    wq[f] = acc;
  }
  __syncthreads();
  for (int o = 0; o < 5; ++o) {
    float p = 0.f;
    for (int f = tid; f < 512; f += 256) p += wq[f] * fcw[o * 512 + f];
    red[tid] = p; __syncthreads();
    for (int s = 128; s > 0; s >>= 1) { if (tid < s) red[tid] += red[tid + s]; __syncthreads(); }
    if (tid == 0) out[b * 5 + o] = red[0] + fcb[o];
    __syncthreads();
  }
}

// ---------------------------------------------------------------------------
// host-side orchestration
// ---------------------------------------------------------------------------
enum {
  IN_INPUTS = 0, IN_RR, IN_RS,
  P_CONV1_W, P_CONV1_B, P_CONV2_W, P_CONV2_B, P_CONV3_W, P_CONV3_B,
  P_BN1_G, P_BN1_B, P_BN2_G, P_BN2_B, P_BN3_G, P_BN3_B, P_BN6_G, P_BN6_B, P_BN7_G, P_BN7_B,
  P_L1F_WIH, P_L1F_WHH, P_L1F_B, P_L1R_WIH, P_L1R_WHH, P_L1R_B,
  P_L2F_WIH, P_L2F_WHH, P_L2F_B, P_L2R_WIH, P_L2R_WHH, P_L2R_B,
  P_MLP1_W, P_MLP1_B, P_MLP1_G, P_MLP1_BETA,
  P_MLP2_W, P_MLP2_B, P_MLP2_G, P_MLP2_BETA,
  P_MLP3_W, P_MLP3_B, P_MLP3_G, P_MLP3_BETA,
  P_OUT_W, P_OUT_B,
  P_MSG1_W, P_MSG1_B, P_MSG2_W, P_MSG2_B,
  P_CONV6_W, P_CONV6_B, P_CONV7_W, P_CONV7_B,
  P_Q_W, P_Q_B, P_BNQ_G, P_BNQ_B,
  P_K_W, P_K_B, P_BNK_G, P_BNK_B,
  P_V_W, P_V_B, P_BNV_G, P_BNV_B,
  P_FC_W, P_FC_B
};

extern "C" void kernel_launch(void* const* d_in, const int* in_sizes, int n_in,
                              void* d_out, int out_size, void* d_ws, size_t ws_size,
                              hipStream_t stream)
{
  (void)in_sizes; (void)n_in; (void)out_size; (void)ws_size;
  char* base = (char*)d_ws;
  auto Hp = [&](size_t o) { return (h16*)(base + o); };
  auto Fq = [&](size_t o) { return (float*)(base + o); };
  auto PF = [&](int i) { return (const float*)d_in[i]; };

  // ---- stats region (zeroed every call) + edge index scratch ----
  float* ST     = Fq(0);
  float* ST_BN1 = ST + 0;    float* ST_BN2 = ST + 128;  float* ST_BN3 = ST + 256;
  float* ST_M1  = ST + 384;  float* ST_M2  = ST + 896;  float* ST_M3  = ST + 1408;
  float* ST_NOD = ST + 1536; float* ST_BN7 = ST + 1540;
  float* ST_K   = ST + 1668; float* ST_V   = ST + 1670; float* ST_Q = ST + 1672;
  int* sendi = (int*)(base + 8192);
  int* reci  = sendi + 20;

  // ---- f16 weight casts (plain [N][K] layouts; bump-allocated, < 2 MB) ----
  size_t off = 16 * 1024;
  auto alloc = [&](size_t bytes) { size_t o = off; off += (bytes + 255) & ~(size_t)255; return o; };
  h16* Wc1   = Hp(alloc(64 * 64 * 2));
  h16* Wc2   = Hp(alloc(64 * 320 * 2));
  h16* Wc3   = Hp(alloc(64 * 320 * 2));
  h16* Wih1f = Hp(alloc(256 * 64 * 2));  h16* Wih1r = Hp(alloc(256 * 64 * 2));
  h16* Whh1f = Hp(alloc(256 * 64 * 2));  h16* Whh1r = Hp(alloc(256 * 64 * 2));
  h16* Wih2f = Hp(alloc(256 * 128 * 2)); h16* Wih2r = Hp(alloc(256 * 128 * 2));
  h16* Whh2f = Hp(alloc(256 * 64 * 2));  h16* Whh2r = Hp(alloc(256 * 64 * 2));
  h16* Wm1   = Hp(alloc(256 * 256 * 2)); h16* Wm2 = Hp(alloc(256 * 256 * 2));
  h16* Wm3   = Hp(alloc(64 * 768 * 2));
  h16* Wg10  = Hp(alloc(256 * 256 * 2)); h16* Wg11 = Hp(alloc(256 * 256 * 2));
  h16* Wg20  = Hp(alloc(256 * 256 * 2)); h16* Wg21 = Hp(alloc(256 * 256 * 2));
  h16* Wc7   = Hp(alloc(64 * 320 * 2));

  // ---- big regions (aggressive reuse; total ws need ~330 MB) ----
  const size_t MB = 1ull << 20;
  const size_t R0 = 4 * MB;        // 128 MB: conv raws / xproj / mergeraw
  const size_t R1 = R0 + 128 * MB; //  64 MB: X1 / X3(time-major) / HB2
  const size_t R2 = R1 + 64 * MB;  //  36 MB: X2 / HB1 / msg buffers
  const size_t R3 = R2 + 36 * MB;  //  96 MB: encoder/decoder small tensors

  h16* Y1  = Hp(R0); h16* Y2 = Hp(R0); h16* Y3 = Hp(R0);
  h16* XPf = Hp(R0); h16* XPr = Hp(R0 + 58 * MB);
  float* mergeraw = Fq(R0);
  h16* X1 = Hp(R1); h16* X3 = Hp(R1); h16* HB2 = Hp(R1);
  h16* X2 = Hp(R2); h16* HB1 = Hp(R2);
  h16* mA = Hp(R2); h16* m20 = Hp(R2 + 12 * MB); h16* m21 = Hp(R2 + 24 * MB);
  h16* EF1   = Hp(R3);
  h16* M1raw = Hp(R3 + 12 * MB);
  h16* M1out = Hp(R3 + 24 * MB);   // x_skip
  h16* N1    = Hp(R3 + 36 * MB);
  h16* M2raw = Hp(R3 + 39 * MB);
  h16* M2out = Hp(R3 + 42 * MB);
  h16* EF3   = Hp(R3 + 45 * MB);   // 30 MB, dead before nod/kraw/vraw reuse it
  float* nod  = Fq(R3 + 45 * MB);
  float* kraw = Fq(R3 + 56 * MB);
  float* vraw = Fq(R3 + 67 * MB);
  float* qraw = Fq(R3 + 78 * MB);
  float* relt = Fq(R3 + 81 * MB);
  h16* M3raw  = Hp(R3 + 82 * MB);
  h16* M3out  = Hp(R3 + 85 * MB);
  float* out = (float*)d_out;

  hipMemsetAsync(base, 0, 8192, stream);
  k_edge_idx<<<1, 32, 0, stream>>>(PF(IN_RR), PF(IN_RS), sendi, reci);

  // plain f32 -> f16 weight casts (layouts already [N][K])
  k_cast_conv1<<<16, 256, 0, stream>>>(Wc1, PF(P_CONV1_W));
  k_cast<<<80, 256, 0, stream>>>(Wc2, PF(P_CONV2_W), 20480);
  k_cast<<<80, 256, 0, stream>>>(Wc3, PF(P_CONV3_W), 20480);
  k_cast<<<64, 256, 0, stream>>>(Wih1f, PF(P_L1F_WIH), 16384);
  k_cast<<<64, 256, 0, stream>>>(Wih1r, PF(P_L1R_WIH), 16384);
  k_cast<<<64, 256, 0, stream>>>(Whh1f, PF(P_L1F_WHH), 16384);
  k_cast<<<64, 256, 0, stream>>>(Whh1r, PF(P_L1R_WHH), 16384);
  k_cast<<<128, 256, 0, stream>>>(Wih2f, PF(P_L2F_WIH), 32768);
  k_cast<<<128, 256, 0, stream>>>(Wih2r, PF(P_L2R_WIH), 32768);
  k_cast<<<64, 256, 0, stream>>>(Whh2f, PF(P_L2F_WHH), 16384);
  k_cast<<<64, 256, 0, stream>>>(Whh2r, PF(P_L2R_WHH), 16384);
  k_cast<<<256, 256, 0, stream>>>(Wm1, PF(P_MLP1_W), 65536);
  k_cast<<<256, 256, 0, stream>>>(Wm2, PF(P_MLP2_W), 65536);
  k_cast<<<192, 256, 0, stream>>>(Wm3, PF(P_MLP3_W), 49152);
  k_cast<<<256, 256, 0, stream>>>(Wg10, PF(P_MSG1_W), 65536);
  k_cast<<<256, 256, 0, stream>>>(Wg11, PF(P_MSG1_W) + 65536, 65536);
  k_cast<<<256, 256, 0, stream>>>(Wg20, PF(P_MSG2_W), 65536);
  k_cast<<<256, 256, 0, stream>>>(Wg21, PF(P_MSG2_W) + 65536, 65536);
  k_cast<<<80, 256, 0, stream>>>(Wc7, PF(P_CONV7_W), 20480);

  // ---- CNN stack (conv biases cancel in training-mode BN) ----
  k_conv1<<<dim3(5120, 3), 256, 0, stream>>>(PF(IN_INPUTS), Wc1, Y1);
  k_stats_ch<h16><<<512, 256, 0, stream>>>(Y1, 5120, 192, ST_BN1);
  k_bn_pool<<<4096, 256, 0, stream>>>(Y1, 5120, 192, ST_BN1, 983040.f,
                                      PF(P_BN1_G), PF(P_BN1_B), X1, 0);
  k_conv23<<<dim3(5120, 2), 256, 0, stream>>>(X1, Wc2, Y2, 96, 92);
  k_stats_ch<h16><<<512, 256, 0, stream>>>(Y2, 5120, 92, ST_BN2);
  k_bn_pool<<<4096, 256, 0, stream>>>(Y2, 5120, 92, ST_BN2, 471040.f,
                                      PF(P_BN2_G), PF(P_BN2_B), X2, 0);
  k_conv23<<<dim3(5120, 1), 256, 0, stream>>>(X2, Wc3, Y3, 46, 42);
  k_stats_ch<h16><<<512, 256, 0, stream>>>(Y3, 5120, 42, ST_BN3);
  k_bn_pool<<<4096, 256, 0, stream>>>(Y3, 5120, 42, ST_BN3, 215040.f,
                                      PF(P_BN3_G), PF(P_BN3_B), X3, 1);  // time-major

  // ---- BiLSTM layer 1: precompute x-projection (bias folded), then scan ----
  k_gemm64<<<dim3(1680, 4), 256, 0, stream>>>(X3, 64, Wih1f, 64, 64, PF(P_L1F_B), 0, XPf, nullptr, 256);
  k_gemm64<<<dim3(1680, 4), 256, 0, stream>>>(X3, 64, Wih1r, 64, 64, PF(P_L1R_B), 0, XPr, nullptr, 256);
  k_lstm<<<160, 256, 0, stream>>>(XPf, Whh1f, HB1, 0, 0);
  k_lstm<<<160, 256, 0, stream>>>(XPr, Whh1r, HB1, 1, 64);
  // ---- BiLSTM layer 2 ----
  k_gemm64<<<dim3(1680, 4), 256, 0, stream>>>(HB1, 128, Wih2f, 128, 128, PF(P_L2F_B), 0, XPf, nullptr, 256);
  k_gemm64<<<dim3(1680, 4), 256, 0, stream>>>(HB1, 128, Wih2r, 128, 128, PF(P_L2R_B), 0, XPr, nullptr, 256);
  k_lstm<<<160, 256, 0, stream>>>(XPf, Whh2f, HB2, 0, 0);
  k_lstm<<<160, 256, 0, stream>>>(XPr, Whh2r, HB2, 1, 64);
  const h16* sinput = HB2 + (size_t)20 * 5120 * 128;  // x[-1] = [5120, 128]

  // ---- NRI encoder (mlp biases cancel in BN) ----
  k_node2edge<<<2048, 256, 0, stream>>>(sinput, 128, sendi, reci, EF1, 256, 0);
  k_gemm64<<<dim3(320, 4), 256, 0, stream>>>(EF1, 256, Wm1, 256, 256, nullptr, 0, M1raw, nullptr, 256);
  k_stats_feat<<<64, 256, 0, stream>>>(M1raw, 20480, 256, ST_M1);
  k_bn_feat<<<2048, 256, 0, stream>>>(M1raw, 20480, 256, ST_M1, 20480.f,
                                      PF(P_MLP1_G), PF(P_MLP1_BETA), M1out);
  k_edge2node<<<1024, 256, 0, stream>>>(M1out, 256, reci, N1);
  k_gemm64<<<dim3(80, 4), 256, 0, stream>>>(N1, 256, Wm2, 256, 256, nullptr, 0, M2raw, nullptr, 256);
  k_stats_feat<<<64, 256, 0, stream>>>(M2raw, 5120, 256, ST_M2);
  k_bn_feat<<<1024, 256, 0, stream>>>(M2raw, 5120, 256, ST_M2, 5120.f,
                                      PF(P_MLP2_G), PF(P_MLP2_BETA), M2out);
  k_node2edge<<<2048, 256, 0, stream>>>(M2out, 256, sendi, reci, EF3, 768, 0);
  k_copy_rows<<<2048, 256, 0, stream>>>(M1out, 256, EF3, 768, 512, 20480);
  k_gemm64<<<dim3(320, 1), 256, 0, stream>>>(EF3, 768, Wm3, 768, 768, nullptr, 0, M3raw, nullptr, 64);
  k_stats_feat<<<64, 256, 0, stream>>>(M3raw, 20480, 64, ST_M3);
  k_bn_feat<<<1024, 256, 0, stream>>>(M3raw, 20480, 64, ST_M3, 20480.f,
                                      PF(P_MLP3_G), PF(P_MLP3_BETA), M3out);
  k_logits<<<80, 256, 0, stream>>>(M3out, PF(P_OUT_W), PF(P_OUT_B), relt);

  // ---- decoder messages (relu MLPs keep their biases) ----
  k_gemm64<<<dim3(320, 4), 256, 0, stream>>>(EF1, 256, Wg10, 256, 256, PF(P_MSG1_B), 1, mA, nullptr, 256);
  k_gemm64<<<dim3(320, 4), 256, 0, stream>>>(mA, 256, Wg20, 256, 256, PF(P_MSG2_B), 1, m20, nullptr, 256);
  k_gemm64<<<dim3(320, 4), 256, 0, stream>>>(EF1, 256, Wg11, 256, 256, PF(P_MSG1_B) + 256, 1, mA, nullptr, 256);
  k_gemm64<<<dim3(320, 4), 256, 0, stream>>>(mA, 256, Wg21, 256, 256, PF(P_MSG2_B) + 256, 1, m21, nullptr, 256);
  k_aggregate<<<2048, 256, 0, stream>>>(m20, m21, relt, reci, nod);

  // ---- conv6(analytic BN6) + conv7 WMMA + fused K/V, then attention ----
  k_stats_scalar<<<512, 256, 0, stream>>>(nod, 2621440l, ST_NOD);
  k_conv7<<<8192, 256, 0, stream>>>(nod, ST_NOD, PF(P_CONV6_W), PF(P_BN6_G), PF(P_BN6_B),
                                    Wc7, PF(P_K_W), PF(P_V_W),
                                    mergeraw, kraw, vraw, 2621440.f);
  k_stats_scalar<<<512, 256, 0, stream>>>(kraw, 2621440l, ST_K);
  k_stats_scalar<<<512, 256, 0, stream>>>(vraw, 2621440l, ST_V);
  k_stats_ch<float><<<512, 256, 0, stream>>>(mergeraw, 1024, 512, ST_BN7);
  k_q<<<1024, 256, 0, stream>>>(mergeraw, ST_BN7, PF(P_BN7_G), PF(P_BN7_B), PF(P_Q_W),
                                qraw, 524288.f);
  k_stats_scalar<<<512, 256, 0, stream>>>(qraw, 524288l, ST_Q);
  k_attn<<<1024, 256, 0, stream>>>(qraw, kraw, vraw, ST_Q, ST_K, ST_V,
                                   PF(P_BNQ_G), PF(P_BNQ_B), PF(P_BNK_G), PF(P_BNK_B),
                                   PF(P_BNV_G), PF(P_BNV_B),
                                   PF(P_FC_W), PF(P_FC_B), out, 524288.f, 2621440.f);
}
